// LSTMRegressor_21010980012295
// MI455X (gfx1250) — compile-verified, matrix operand order checked by reading, not run on hardware
//
#include <hip/hip_runtime.h>
#include <hip/hip_bf16.h>
#include <stdint.h>

// ---------------------------------------------------------------------------
// Persistent fused 2-layer LSTM for MI455X (gfx1250, wave32, WMMA fp8).
//  - 16 workgroups x 512 threads (16 waves). Each WG owns 16 batch rows.
//  - W_hh (fp8, 256KB) staged into LDS by the Tensor Data Mover, resident
//    for the whole 512-step scan.
//  - Wave w owns hidden cols [16w,16w+16): computes i/f/g/o tiles for those
//    cols -> elementwise LSTM update fully in registers (c = one v8f).
//  - h / staged inputs stored in *fragment-permuted* order so every WMMA
//    operand is a contiguous 64B per-lane load (ds_load_b128 / b128 global).
//  - All f32->fp8 conversion via v_cvt_pk_fp8_f32 when available.
// ---------------------------------------------------------------------------

typedef __attribute__((ext_vector_type(8)))  int      v8i;
typedef __attribute__((ext_vector_type(16))) int      v16i;
typedef __attribute__((ext_vector_type(8)))  float    v8f;
typedef __attribute__((ext_vector_type(4)))  unsigned u32x4;
typedef __attribute__((ext_vector_type(8)))  int      i32x8;
typedef __attribute__((ext_vector_type(4)))  int      i32x4;

#define HID   256
#define T_LEN 512
#define B_SZ  256
#define BLK   512

// ws layout (bytes)
#define OFF_WIH0  0u          //  64KB  fp8 fragments, 16x16x64 layout
#define OFF_WHH0  65536u      // 256KB  fp8 fragments, 16x16x128 layout
#define OFF_WIH1  327680u     // 256KB
#define OFF_WHH1  589824u     // 256KB
#define OFF_B0    851968u     //   4KB  f32 combined bias layer0
#define OFF_B1    856064u     //   4KB
#define OFF_OUT0  860160u     // 33.5MB fp8 h0[b][t][256] (fragment-permuted rows)

#if __has_builtin(__builtin_amdgcn_tensor_load_to_lds)
#define HAVE_TDM 1
#else
#define HAVE_TDM 0
#endif

#if __has_builtin(__builtin_amdgcn_cvt_pk_fp8_f32)
#define HAVE_CVT_FP8 1
#else
#define HAVE_CVT_FP8 0
#endif

// ---- fallback fp32 -> fp8 E4M3 (RNE-ish, saturating) ----------------------
__device__ inline unsigned f32_to_e4m3(float f) {
  unsigned u = __float_as_uint(f);
  unsigned s = (u >> 24) & 0x80u;
  unsigned a = u & 0x7FFFFFFFu;
  if (a >= 0x43E00000u) return s | 0x7Eu;          // >=448 or NaN -> clamp
  int e = (int)(a >> 23) - 120;                    // biased target exponent
  unsigned mant = (a >> 20) & 7u;
  unsigned rest = a & 0xFFFFFu;
  if (e <= 0) {                                    // subnormal / underflow
    if (e < -3 || a == 0u) return s;
    unsigned full = 8u | mant;
    int sh = 1 - e;
    unsigned q = (full >> sh) + ((full >> (sh - 1)) & 1u);
    return (q > 7u) ? (s | 0x08u) : (s | q);
  }
  if (rest > 0x80000u || (rest == 0x80000u && (mant & 1u))) {
    if (++mant == 8u) { mant = 0u; ++e; }
  }
  if (e >= 16 || (e == 15 && mant == 7u)) return s | 0x7Eu;
  return s | ((unsigned)e << 3) | mant;
}

// Packed fp8 converts: 2 floats -> bits[15:0]; 4 floats -> full dword.
__device__ inline unsigned pk2_fp8(float a, float b) {
#if HAVE_CVT_FP8
  return (unsigned)__builtin_amdgcn_cvt_pk_fp8_f32(a, b, 0, false) & 0xFFFFu;
#else
  return f32_to_e4m3(a) | (f32_to_e4m3(b) << 8);
#endif
}
__device__ inline unsigned pk4_fp8(float a, float b, float c, float d) {
#if HAVE_CVT_FP8
  int r = __builtin_amdgcn_cvt_pk_fp8_f32(a, b, 0, false);
  r = __builtin_amdgcn_cvt_pk_fp8_f32(c, d, r, true);
  return (unsigned)r;
#else
  return f32_to_e4m3(a) | (f32_to_e4m3(b) << 8) |
         (f32_to_e4m3(c) << 16) | (f32_to_e4m3(d) << 24);
#endif
}

// K-byte position of dword v (0..7) of the 8-bit A/B 16x64 fragment layout.
__device__ inline int kl64(int v, int lane) {
  return ((v >> 1) << 4) + ((v & 1) << 2) + ((lane & 16) ? 8 : 0);
}

// Fragment-permuted dword position for K=256 rows: consumer lane (lo/hi, kt)
// reads 16 *contiguous* dwords at (hi?32:0)+kt*16.
__device__ inline int posdw256(int kdw) {
  return ((kdw & 2) ? 32 : 0) + ((kdw >> 2) << 1) + (kdw & 1);
}
// Same for K=64 rows: consumer reads 8 contiguous dwords at (hi?8:0).
__device__ inline int posdw64(int kdw) {
  return ((kdw & 2) ? 8 : 0) + ((kdw >> 2) << 1) + (kdw & 1);
}

__device__ inline float sigf(float x) { return 1.0f / (1.0f + __expf(-x)); }

// ---- Tensor Data Mover: 1-row 2D descriptor, data_size=8B ------------------
__device__ inline void tdm_load_to_lds(const void* gptr, unsigned lds_off,
                                       unsigned bytes) {
#if HAVE_TDM
  unsigned long long ga = (unsigned long long)gptr;
  unsigned units = bytes >> 3;                     // 8-byte elements
  u32x4 g0;
  g0[0] = 1u;                                      // count=1 (user D#)
  g0[1] = lds_off;                                 // lds_addr (bytes)
  g0[2] = (unsigned)ga;                            // global_addr[31:0]
  g0[3] = (unsigned)((ga >> 32) & 0x01FFFFFFull) | 0x80000000u; // [56:32]|type=2
  i32x8 g1;
  g1[0] = 0x30000;                                 // data_size=3 (8B)
  g1[1] = (int)(units << 16);                      // tensor_dim0[15:0]
  g1[2] = (int)((units >> 16) | (1u << 16));       // tensor_dim0[31:16]|dim1=1
  g1[3] = (int)(units << 16);                      // tile_dim0 (fits 16b: 32768)
  g1[4] = 1;                                       // tile_dim1=1
  g1[5] = (int)units;                              // tensor_dim0_stride lo32
  g1[6] = 0;
  g1[7] = 0;
  i32x4 z4 = {};
#if defined(__clang_major__) && (__clang_major__ >= 23)
  i32x8 z8 = {};
  __builtin_amdgcn_tensor_load_to_lds(g0, g1, z4, z4, z8, 0);
#else
  __builtin_amdgcn_tensor_load_to_lds(g0, g1, z4, z4, 0);
#endif
#else
  (void)gptr; (void)lds_off; (void)bytes;
#endif
}

// ---------------------------------------------------------------------------
// Pack kernels: f32 [4H x K] row-major -> fp8 B-operand fragment streams.
// ---------------------------------------------------------------------------
__global__ void pack_x128(const float* __restrict__ W, unsigned* __restrict__ dst) {
  int idx  = blockIdx.x * 256 + threadIdx.x;       // 65536 dwords total
  int tile = idx >> 10;
  int rem  = idx & 1023;
  int kt   = rem >> 9;
  int r2   = rem & 511;
  int lane = r2 >> 4;
  int v    = r2 & 15;
  int n    = tile * 16 + (lane & 15);
  int k    = kt * 128 + ((v & 8) ? 64 : 0) + kl64(v & 7, lane);
  const float* w = W + (size_t)n * 256 + k;
  dst[idx] = pk4_fp8(w[0], w[1], w[2], w[3]);
}

__global__ void pack_x64(const float* __restrict__ W, unsigned* __restrict__ dst) {
  int idx  = blockIdx.x * 256 + threadIdx.x;       // 16384 dwords total
  int tile = idx >> 8;
  int lane = (idx >> 3) & 31;
  int v    = idx & 7;
  int n    = tile * 16 + (lane & 15);
  int k    = kl64(v, lane);
  const float* w = W + (size_t)n * 64 + k;
  dst[idx] = pk4_fp8(w[0], w[1], w[2], w[3]);
}

__global__ void pack_bias(const float* bi0, const float* bh0,
                          const float* bi1, const float* bh1,
                          float* o0, float* o1) {
  int i = blockIdx.x * 256 + threadIdx.x;          // 2048
  if (i < 1024) o0[i] = bi0[i] + bh0[i];
  else          o1[i - 1024] = bi1[i - 1024] + bh1[i - 1024];
}

// ---------------------------------------------------------------------------
// Persistent LSTM kernel. Dynamic LDS:
//   [0,256K)      s_w  : W_hh fragments of current layer (TDM-staged)
//   [256K,+4K)    s_h  : h fp8, fragment-permuted (u32[16][64])
//   [260K,+4K)    s_x  : staged input tile fp8, fragment-permuted
//   [264K,+16K)   s_hf : final h1 in f32 for the head
// ---------------------------------------------------------------------------
__global__ __launch_bounds__(BLK, 1)
void lstm_persist(const float* __restrict__ x,
                  const unsigned* __restrict__ pWih0, const unsigned* __restrict__ pWhh0,
                  const unsigned* __restrict__ pWih1, const unsigned* __restrict__ pWhh1,
                  const float* __restrict__ bias0, const float* __restrict__ bias1,
                  const float* __restrict__ Whead, const float* __restrict__ bhead,
                  unsigned char* __restrict__ out0, float* __restrict__ out) {
  extern __shared__ unsigned char smem[];
  unsigned* s_w  = (unsigned*)smem;
  unsigned* s_h  = (unsigned*)(smem + 262144);
  unsigned* s_x  = (unsigned*)(smem + 262144 + 4096);
  float*    s_hf = (float*)(smem + 262144 + 8192);

  const int tid  = threadIdx.x;
  const int wave = tid >> 5;
  const int lane = tid & 31;
  const int bT   = blockIdx.x;
  const int m    = lane & 15;                 // A-operand row for this lane
  const int n    = wave * 16 + (lane & 15);   // hidden column this lane updates
  const int rOff = (lane & 16) ? 8 : 0;       // C-tile row offset
  const int hByteOff = posdw256(n >> 2) * 4 + (n & 3); // permuted h byte slot
  const int aHi  = (lane & 16) ? 32 : 0;      // A-frag base (K=256 rows)
  const int aHi64 = (lane & 16) ? 8 : 0;      // A-frag base (K=64 rows)

  // ================= Phase A: layer 0 =================
#if HAVE_TDM
  if (wave == 0) {
    tdm_load_to_lds(pWhh0, 0u, 262144u);
    __builtin_amdgcn_s_wait_tensorcnt(0);
  }
#else
  for (int i = tid; i < 65536 / 4; i += BLK)
    ((uint4*)s_w)[i] = ((const uint4*)pWhh0)[i];
#endif
  for (int i = tid; i < 1024; i += BLK) s_h[i] = 0u;

  float bv[4];
#pragma unroll
  for (int g = 0; g < 4; ++g) bv[g] = bias0[g * 256 + n];
  v8f cst = {};
  __syncthreads();

  for (int t = 0; t < T_LEN; ++t) {
    if (tid < 256) {                          // stage x_t -> fp8 (permuted)
      int mm = tid >> 4, q = tid & 15;
      const float* xp = x + (((size_t)(bT * 16 + mm) * T_LEN + t) * 64) + q * 4;
      s_x[mm * 64 + posdw64(q)] = pk4_fp8(xp[0], xp[1], xp[2], xp[3]);
      if (t + 1 < T_LEN) __builtin_prefetch(xp + 64, 0, 0);
    }
    __syncthreads();

    v8i  ax  = *(const v8i*)(s_x + m * 64 + aHi64);
    v16i ah0 = *(const v16i*)(s_h + m * 64 + aHi);
    v16i ah1 = *(const v16i*)(s_h + m * 64 + aHi + 16);

    v8f acc[4];
#pragma unroll
    for (int g = 0; g < 4; ++g) {
      v8f a;
#pragma unroll
      for (int r = 0; r < 8; ++r) a[r] = bv[g];
      v8i bw = *(const v8i*)(pWih0 + (size_t)(g * 16 + wave) * 256 + lane * 8);
      a = __builtin_amdgcn_wmma_f32_16x16x64_fp8_fp8(ax, bw, (short)0, a, false, false);
      v16i bh0 = *(const v16i*)(s_w + ((g * 16 + wave) * 2 + 0) * 512 + lane * 16);
      a = __builtin_amdgcn_wmma_f32_16x16x128_fp8_fp8(ah0, bh0, (short)0, a, false, false);
      v16i bh1 = *(const v16i*)(s_w + ((g * 16 + wave) * 2 + 1) * 512 + lane * 16);
      a = __builtin_amdgcn_wmma_f32_16x16x128_fp8_fp8(ah1, bh1, (short)0, a, false, false);
      acc[g] = a;
    }
    __syncthreads();                          // all s_h reads done

#pragma unroll
    for (int r = 0; r < 8; r += 2) {
      float i0 = sigf(acc[0][r]),     f0 = sigf(acc[1][r]);
      float g0 = tanhf(acc[2][r]),    o0 = sigf(acc[3][r]);
      float i1 = sigf(acc[0][r + 1]), f1 = sigf(acc[1][r + 1]);
      float g1 = tanhf(acc[2][r + 1]), o1 = sigf(acc[3][r + 1]);
      float c0 = f0 * cst[r] + i0 * g0;
      float c1 = f1 * cst[r + 1] + i1 * g1;
      cst[r] = c0; cst[r + 1] = c1;
      float h0 = o0 * tanhf(c0);
      float h1 = o1 * tanhf(c1);
      unsigned p = pk2_fp8(h0, h1);
      int row0 = r + rOff, row1 = r + 1 + rOff;
      ((unsigned char*)s_h)[row0 * 256 + hByteOff] = (unsigned char)(p & 0xFF);
      ((unsigned char*)s_h)[row1 * 256 + hByteOff] = (unsigned char)(p >> 8);
      out0[(((size_t)(bT * 16 + row0) * T_LEN + t) * 256) + hByteOff] = (unsigned char)(p & 0xFF);
      out0[(((size_t)(bT * 16 + row1) * T_LEN + t) * 256) + hByteOff] = (unsigned char)(p >> 8);
    }
    __syncthreads();
  }

  __threadfence();                            // out0 stores visible for phase B
  __syncthreads();

  // ================= Phase B: layer 1 =================
#if HAVE_TDM
  if (wave == 0) {
    tdm_load_to_lds(pWhh1, 0u, 262144u);
    __builtin_amdgcn_s_wait_tensorcnt(0);
  }
#else
  for (int i = tid; i < 65536 / 4; i += BLK)
    ((uint4*)s_w)[i] = ((const uint4*)pWhh1)[i];
#endif
  for (int i = tid; i < 1024; i += BLK) s_h[i] = 0u;
#pragma unroll
  for (int g = 0; g < 4; ++g) bv[g] = bias1[g * 256 + n];
#pragma unroll
  for (int r = 0; r < 8; ++r) cst[r] = 0.0f;
  __syncthreads();

  for (int t = 0; t < T_LEN; ++t) {
    for (int i = tid; i < 1024; i += BLK) {   // stage h0_t (already permuted)
      int mm = i >> 6, q = i & 63;
      const unsigned* src =
          (const unsigned*)out0 +
          ((((size_t)(bT * 16 + mm) * T_LEN + t) * 256) >> 2) + q;
      s_x[i] = *src;
      if (t + 1 < T_LEN) __builtin_prefetch(src + 64, 0, 0);
    }
    __syncthreads();

    v16i ai0 = *(const v16i*)(s_x + m * 64 + aHi);
    v16i ai1 = *(const v16i*)(s_x + m * 64 + aHi + 16);
    v16i ah0 = *(const v16i*)(s_h + m * 64 + aHi);
    v16i ah1 = *(const v16i*)(s_h + m * 64 + aHi + 16);

    v8f acc[4];
#pragma unroll
    for (int g = 0; g < 4; ++g) {
      v8f a;
#pragma unroll
      for (int r = 0; r < 8; ++r) a[r] = bv[g];
      // input GEMM: W_ih1 fragments streamed from L2 (b128 per lane)
      v16i bw0 = *(const v16i*)(pWih1 + (size_t)((g * 16 + wave) * 2 + 0) * 512 + lane * 16);
      a = __builtin_amdgcn_wmma_f32_16x16x128_fp8_fp8(ai0, bw0, (short)0, a, false, false);
      v16i bw1 = *(const v16i*)(pWih1 + (size_t)((g * 16 + wave) * 2 + 1) * 512 + lane * 16);
      a = __builtin_amdgcn_wmma_f32_16x16x128_fp8_fp8(ai1, bw1, (short)0, a, false, false);
      // recurrent GEMM: W_hh1 from LDS
      v16i bh0 = *(const v16i*)(s_w + ((g * 16 + wave) * 2 + 0) * 512 + lane * 16);
      a = __builtin_amdgcn_wmma_f32_16x16x128_fp8_fp8(ah0, bh0, (short)0, a, false, false);
      v16i bh1 = *(const v16i*)(s_w + ((g * 16 + wave) * 2 + 1) * 512 + lane * 16);
      a = __builtin_amdgcn_wmma_f32_16x16x128_fp8_fp8(ah1, bh1, (short)0, a, false, false);
      acc[g] = a;
    }
    __syncthreads();

#pragma unroll
    for (int r = 0; r < 8; r += 2) {
      float i0 = sigf(acc[0][r]),      f0 = sigf(acc[1][r]);
      float g0 = tanhf(acc[2][r]),     o0 = sigf(acc[3][r]);
      float i1 = sigf(acc[0][r + 1]),  f1 = sigf(acc[1][r + 1]);
      float g1 = tanhf(acc[2][r + 1]), o1 = sigf(acc[3][r + 1]);
      float c0 = f0 * cst[r] + i0 * g0;
      float c1 = f1 * cst[r + 1] + i1 * g1;
      cst[r] = c0; cst[r + 1] = c1;
      float h0 = o0 * tanhf(c0);
      float h1 = o1 * tanhf(c1);
      unsigned p = pk2_fp8(h0, h1);
      int row0 = r + rOff, row1 = r + 1 + rOff;
      ((unsigned char*)s_h)[row0 * 256 + hByteOff] = (unsigned char)(p & 0xFF);
      ((unsigned char*)s_h)[row1 * 256 + hByteOff] = (unsigned char)(p >> 8);
      if (t == T_LEN - 1) {
        s_hf[row0 * 256 + n] = h0;
        s_hf[row1 * 256 + n] = h1;
      }
    }
    __syncthreads();
  }

  // ================= Phase C: linear head =================
  if (tid < 16) {
    float s = bhead[0];
    for (int k = 0; k < 256; ++k) s += s_hf[tid * 256 + k] * Whead[k];
    out[bT * 16 + tid] = s;
  }
}

// ---------------------------------------------------------------------------
extern "C" void kernel_launch(void* const* d_in, const int* in_sizes, int n_in,
                              void* d_out, int out_size, void* d_ws, size_t ws_size,
                              hipStream_t stream) {
  (void)in_sizes; (void)n_in; (void)out_size; (void)ws_size;
  const float* x     = (const float*)d_in[0];
  const float* Wih0  = (const float*)d_in[1];
  const float* Whh0  = (const float*)d_in[2];
  const float* bih0  = (const float*)d_in[3];
  const float* bhh0  = (const float*)d_in[4];
  const float* Wih1  = (const float*)d_in[5];
  const float* Whh1  = (const float*)d_in[6];
  const float* bih1  = (const float*)d_in[7];
  const float* bhh1  = (const float*)d_in[8];
  const float* Whead = (const float*)d_in[9];
  const float* bhead = (const float*)d_in[10];

  uint8_t* ws = (uint8_t*)d_ws;
  unsigned* pWih0 = (unsigned*)(ws + OFF_WIH0);
  unsigned* pWhh0 = (unsigned*)(ws + OFF_WHH0);
  unsigned* pWih1 = (unsigned*)(ws + OFF_WIH1);
  unsigned* pWhh1 = (unsigned*)(ws + OFF_WHH1);
  float*    b0    = (float*)(ws + OFF_B0);
  float*    b1    = (float*)(ws + OFF_B1);
  unsigned char* o0 = ws + OFF_OUT0;

  pack_x64 <<<64,  256, 0, stream>>>(Wih0, pWih0);
  pack_x128<<<256, 256, 0, stream>>>(Whh0, pWhh0);
  pack_x128<<<256, 256, 0, stream>>>(Wih1, pWih1);
  pack_x128<<<256, 256, 0, stream>>>(Whh1, pWhh1);
  pack_bias<<<8,   256, 0, stream>>>(bih0, bhh0, bih1, bhh1, b0, b1);

  // 256K (W_hh) + 4K (h fp8) + 4K (input stage) + 16K (h1 f32) = 280KB LDS
  lstm_persist<<<B_SZ / 16, BLK, 286720, stream>>>(
      x, pWih0, pWhh0, pWih1, pWhh1, b0, b1, Whead, bhead, o0, (float*)d_out);
}
